// C3Net_Interaction_30623116820559
// MI455X (gfx1250) — matrix operand order repaired and checked by reading, not compiled, not run on hardware
//
#include <hip/hip_runtime.h>
#include <hip/hip_bf16.h>
#include <math.h>

// ---------------------------------------------------------------------------
// CDNA5 (gfx1250) wave32 WMMA types & helpers
// ---------------------------------------------------------------------------
typedef __attribute__((ext_vector_type(16))) __bf16 v16bf;
typedef __attribute__((ext_vector_type(8)))  __bf16 v8bf;
typedef __attribute__((ext_vector_type(8)))  float  v8f;
typedef __attribute__((ext_vector_type(4)))  unsigned int v4u;
typedef __attribute__((ext_vector_type(8)))  int    v8i;
typedef __attribute__((ext_vector_type(4)))  int    v4i;

#if __has_builtin(__builtin_amdgcn_tensor_load_to_lds)
#define HAVE_TDM 1
#else
#define HAVE_TDM 0
#endif

#define LOG2F_ 0.693147180559945f

__device__ __forceinline__ float ssp_f(float x) {
  float sp = (x > 20.0f) ? x : log1pf(__expf(x));
  return sp - LOG2F_;
}

// A fragment (16Mx32K bf16) from row-major LDS tile [m][k], rowStride in elems.
__device__ __forceinline__ v16bf load_frag_A(const __bf16* base, int lane, int rowStride) {
  const int m  = lane & 15;
  const int kh = (lane >> 4) & 1;
  const __bf16* p = base + m * rowStride + kh * 8;
  v8bf lo = *(const v8bf*)p;          // K = kh*8 ..
  v8bf hi = *(const v8bf*)(p + 16);   // K = 16 + kh*8 ..
  return __builtin_shufflevector(lo, hi, 0,1,2,3,4,5,6,7,8,9,10,11,12,13,14,15);
}

// B fragment (32Kx16N bf16) from LDS stored TRANSPOSED as [n][k].
__device__ __forceinline__ v16bf load_frag_B(const __bf16* base, int lane, int rowStride) {
  const int n  = lane & 15;
  const int kh = (lane >> 4) & 1;
  const __bf16* p = base + n * rowStride + kh * 8;
  v8bf lo = *(const v8bf*)p;
  v8bf hi = *(const v8bf*)(p + 16);
  return __builtin_shufflevector(lo, hi, 0,1,2,3,4,5,6,7,8,9,10,11,12,13,14,15);
}

__device__ __forceinline__ v8f wmma_bf16(v16bf a, v16bf b, v8f c) {
  return __builtin_amdgcn_wmma_f32_16x16x32_bf16(false, a, false, b, (short)0, c, false, false);
}

__device__ __forceinline__ int c_row(int r, int lane) { return r + 8 * ((lane >> 4) & 1); }

// ---------------------------------------------------------------------------
// TDM: async 2D f32 tile load Global -> LDS via Tensor Data Mover.
// Builds a D# (groups 0/1; groups 2/3 zero => 2D tensor) and issues
// tensor_load_to_lds. Completion via TENSORcnt.
// ---------------------------------------------------------------------------
#if HAVE_TDM
__device__ __forceinline__ void tdm_load_2d_f32(void* lds_dst, const void* gsrc,
                                                int tile_w, int tile_h, int stride_elems) {
  unsigned long long ga = (unsigned long long)(uintptr_t)gsrc;
  v4u g0;
  g0[0] = 1u;                                            // count=1 valid user D#
  g0[1] = (unsigned)(uintptr_t)lds_dst;                  // lds_addr (bytes)
  g0[2] = (unsigned)(ga & 0xffffffffu);                  // global_addr lo
  g0[3] = (unsigned)((ga >> 32) & 0x1ffffffu) | (2u << 30); // global_addr hi | type=2
  v8i g1;
  g1[0] = (2 << 16);                                     // data_size = 4 bytes
  g1[1] = (tile_w & 0xffff) << 16;                       // tensor_dim0[15:0]
  g1[2] = (tile_h & 0xffff) << 16;                       // tensor_dim0 hi=0 | tensor_dim1[15:0]
  g1[3] = (tile_w & 0xffff) << 16;                       // tensor_dim1 hi=0 | tile_dim0
  g1[4] = (tile_h & 0xffff);                             // tile_dim1 | tile_dim2=0
  g1[5] = stride_elems;                                  // tensor_dim0_stride lo32
  g1[6] = 0;
  g1[7] = 0;
  v4i gz = {0, 0, 0, 0};
#if __clang_major__ >= 23
  v8i gz8 = {0, 0, 0, 0, 0, 0, 0, 0};
  __builtin_amdgcn_tensor_load_to_lds(g0, g1, gz, gz, gz8, 0);
#else
  __builtin_amdgcn_tensor_load_to_lds(g0, g1, gz, gz, 0);
#endif
}
#endif

// ---------------------------------------------------------------------------
// Kernel 0: one-shot fp32 -> bf16 weight conversion into transposed (N-major)
// layouts so workgroup LDS fills become straight vector copies.
// ---------------------------------------------------------------------------
__global__ void k_prep(const float* __restrict__ Wf1, const float* __restrict__ Wf2,
                       const float* __restrict__ Win2f, const float* __restrict__ Wf2out,
                       const float* __restrict__ Wdense,
                       __bf16* __restrict__ W1t, __bf16* __restrict__ W2t,
                       __bf16* __restrict__ Wint, __bf16* __restrict__ Wot,
                       __bf16* __restrict__ Wdt) {
  const int t0 = blockIdx.x * blockDim.x + threadIdx.x;
  const int stride = gridDim.x * blockDim.x;
  for (int e = t0; e < 128 * 64; e += stride) {      // [f][s], S padded 50->64
    int si = e & 63, f = e >> 6;
    W1t[e] = (si < 50) ? (__bf16)Wf1[si * 128 + f] : (__bf16)0.0f;
  }
  for (int e = t0; e < 128 * 128; e += stride) {     // [n][k]
    int k = e & 127, n = e >> 7;
    W2t[e]  = (__bf16)Wf2[k * 128 + n];
    Wint[e] = (__bf16)Win2f[k * 128 + n];
    Wot[e]  = (__bf16)Wf2out[k * 128 + n];
    Wdt[e]  = (__bf16)Wdense[k * 128 + n];
  }
}

// ---------------------------------------------------------------------------
// Kernel 1: y[8192,128] = s[8192,128] @ W_in2f[128,128]   (bf16 WMMA)
// ---------------------------------------------------------------------------
__global__ void k_in2f(const float* __restrict__ s, const __bf16* __restrict__ Wt,
                       float* __restrict__ y) {
  extern __shared__ char smem[];
  __bf16* sA  = (__bf16*)smem;                 // 16 x 128
  __bf16* sBt = (__bf16*)(smem + 4096);        // 128(n) x 128(k)
  const int tid = threadIdx.x, lane = tid & 31, wave = tid >> 5;
  const int m0 = blockIdx.x * 16;

  const uint4* src = (const uint4*)Wt;
  uint4* dst = (uint4*)sBt;
  for (int e = tid; e < 2048; e += 256) dst[e] = src[e];   // bf16 copy, no cvt
  for (int e = tid; e < 16 * 128; e += 256) {
    int k = e & 127, r = e >> 7;
    sA[r * 128 + k] = (__bf16)s[(size_t)(m0 + r) * 128 + k];
  }
  __syncthreads();

  const int n0 = wave * 16;
  v8f acc = {};
  for (int k0 = 0; k0 < 4; ++k0) {
    v16bf a = load_frag_A(sA + k0 * 32, lane, 128);
    v16bf b = load_frag_B(sBt + n0 * 128 + k0 * 32, lane, 128);
    acc = wmma_bf16(a, b, acc);
  }
  const int col = n0 + (lane & 15);
  for (int r = 0; r < 8; ++r)
    y[(size_t)(m0 + c_row(r, lane)) * 128 + col] = acc[r];
}

// ---------------------------------------------------------------------------
// Kernel 2: fused edge filter network; 4 (b,n) nodes per workgroup.
// Per node: TDM-load f_ij tile -> LDS, H = ssp(f_ij@W_f1+b), Wfil = H@W_f2+b,
// agg = sum_m mask*Wfil*y_gather — all in LDS, Wfil never touches HBM.
// ---------------------------------------------------------------------------
#define NODES_PER_WG 4

__global__ void k_filter(const float* __restrict__ fij,
                         const __bf16* __restrict__ W1t, const float* __restrict__ bf1,
                         const __bf16* __restrict__ W2t, const float* __restrict__ bf2,
                         const int*   __restrict__ nbr, const float* __restrict__ mask,
                         const float* __restrict__ y,   float* __restrict__ agg) {
  extern __shared__ char smem[];
  __bf16* sW1    = (__bf16*)(smem);              // 128(f) x 64(s)   16384 B
  __bf16* sW2    = (__bf16*)(smem + 16384);      // 128(g) x 128(f)  32768 B
  __bf16* sF     = (__bf16*)(smem + 49152);      // 48 x 64           6144 B
  __bf16* sH     = (__bf16*)(smem + 55296);      // 48 x 128         12288 B
  float*  sY     = (float*) (smem + 67584);      // 48 x 128         24576 B
  float*  sStage = (float*) (smem + 92160);      // 48 x 50 f32       9600 B (TDM dest)
  float*  sM     = (float*) (smem + 101760);     // 48                 192 B
  float*  sAg    = (float*) (smem + 101952);     // 128                512 B
  const int tid = threadIdx.x, lane = tid & 31, wave = tid >> 5;

  __builtin_prefetch(W1t, 0, 1);                 // global_prefetch_b8
  __builtin_prefetch(W2t, 0, 1);

  // Pre-converted bf16 weights -> LDS (straight vector copies)
  {
    const uint4* s1 = (const uint4*)W1t;  uint4* d1 = (uint4*)sW1;
    for (int e = tid; e < 1024; e += 256) d1[e] = s1[e];
    const uint4* s2 = (const uint4*)W2t;  uint4* d2 = (uint4*)sW2;
    for (int e = tid; e < 2048; e += 256) d2[e] = s2[e];
  }

  const int n0  = wave * 16;
  const int col = n0 + (lane & 15);
  const float bias1w = bf1[col];
  const float bias2w = bf2[col];

  for (int node = 0; node < NODES_PER_WG; ++node) {
    const int bn = blockIdx.x * NODES_PER_WG + node;   // = b*256 + n
    const int bbase = bn & ~255;                       // = b*256

#if HAVE_TDM
    if (wave == 0) {   // one TDM op per workgroup-node; EXEC-independent
      tdm_load_2d_f32(sStage, fij + (size_t)bn * 48 * 50, 50, 48, 50);
      __builtin_amdgcn_s_wait_tensorcnt(0);
    }
#endif
    for (int m = tid; m < 48; m += 256) sM[m] = mask[(size_t)bn * 48 + m];
    for (int e = tid; e < 48 * 128; e += 256) {        // gather y[b, nbr[m], :]
      int c = e & 127, m = e >> 7;
      int j = nbr[(size_t)bn * 48 + m];
      sY[m * 128 + c] = y[(size_t)(bbase + j) * 128 + c];
    }
    for (int c = tid; c < 128; c += 256) sAg[c] = 0.0f;
#if HAVE_TDM
    __syncthreads();   // TDM data visible workgroup-wide
    for (int e = tid; e < 48 * 64; e += 256) {         // cvt staged f32 -> bf16, pad K
      int si = e & 63, m = e >> 6;
      sF[m * 64 + si] = (si < 50) ? (__bf16)sStage[m * 50 + si] : (__bf16)0.0f;
    }
#else
    for (int e = tid; e < 48 * 64; e += 256) {
      int si = e & 63, m = e >> 6;
      sF[m * 64 + si] = (si < 50) ? (__bf16)fij[((size_t)bn * 48 + m) * 50 + si]
                                  : (__bf16)0.0f;
    }
#endif
    __syncthreads();

    // GEMM1 (48x64 @ 64x128) + bias + ssp -> sH (bf16)
    for (int mt = 0; mt < 3; ++mt) {
      v8f acc = {};
      for (int k0 = 0; k0 < 2; ++k0) {
        v16bf a = load_frag_A(sF + mt * 16 * 64 + k0 * 32, lane, 64);
        v16bf b = load_frag_B(sW1 + n0 * 64 + k0 * 32, lane, 64);
        acc = wmma_bf16(a, b, acc);
      }
      for (int r = 0; r < 8; ++r) {
        int row = mt * 16 + c_row(r, lane);
        sH[row * 128 + col] = (__bf16)ssp_f(acc[r] + bias1w);
      }
    }
    __syncthreads();

    // GEMM2 (48x128 @ 128x128) + bias, masked dot with gathered y, reduce over m
    float colacc = 0.0f;
    for (int mt = 0; mt < 3; ++mt) {
      v8f acc = {};
      for (int k0 = 0; k0 < 4; ++k0) {
        v16bf a = load_frag_A(sH + mt * 16 * 128 + k0 * 32, lane, 128);
        v16bf b = load_frag_B(sW2 + n0 * 128 + k0 * 32, lane, 128);
        acc = wmma_bf16(a, b, acc);
      }
      for (int r = 0; r < 8; ++r) {
        int m = mt * 16 + c_row(r, lane);
        float w = (acc[r] + bias2w) * sM[m];
        colacc += w * sY[m * 128 + col];
      }
    }
    atomicAdd(&sAg[col], colacc);   // ds_add_f32
    __syncthreads();
    for (int c = tid; c < 128; c += 256) agg[(size_t)bn * 128 + c] = sAg[c];
    __syncthreads();                // protect LDS reuse by next node
  }
}

// ---------------------------------------------------------------------------
// Kernel 3: out = ssp(agg @ W_f2out + b_f2out) @ W_dense + b_dense  (fused)
// ---------------------------------------------------------------------------
__global__ void k_out(const float* __restrict__ agg,
                      const __bf16* __restrict__ Wot, const float* __restrict__ bf2out,
                      const __bf16* __restrict__ Wdt, const float* __restrict__ bdense,
                      float* __restrict__ out) {
  extern __shared__ char smem[];
  __bf16* sA  = (__bf16*)(smem);               // 16 x 128
  __bf16* sV  = (__bf16*)(smem + 4096);        // 16 x 128
  __bf16* sB1 = (__bf16*)(smem + 8192);        // 128 x 128
  __bf16* sB2 = (__bf16*)(smem + 8192 + 32768);
  const int tid = threadIdx.x, lane = tid & 31, wave = tid >> 5;
  const int m0 = blockIdx.x * 16;

  {
    const uint4* s1 = (const uint4*)Wot;  uint4* d1 = (uint4*)sB1;
    const uint4* s2 = (const uint4*)Wdt;  uint4* d2 = (uint4*)sB2;
    for (int e = tid; e < 2048; e += 256) { d1[e] = s1[e]; d2[e] = s2[e]; }
  }
  for (int e = tid; e < 16 * 128; e += 256) {
    int k = e & 127, r = e >> 7;
    sA[r * 128 + k] = (__bf16)agg[(size_t)(m0 + r) * 128 + k];
  }
  __syncthreads();

  const int n0  = wave * 16;
  const int col = n0 + (lane & 15);

  v8f acc = {};
  for (int k0 = 0; k0 < 4; ++k0) {
    v16bf a = load_frag_A(sA + k0 * 32, lane, 128);
    v16bf b = load_frag_B(sB1 + n0 * 128 + k0 * 32, lane, 128);
    acc = wmma_bf16(a, b, acc);
  }
  float bias1 = bf2out[col];
  for (int r = 0; r < 8; ++r)
    sV[c_row(r, lane) * 128 + col] = (__bf16)ssp_f(acc[r] + bias1);
  __syncthreads();

  v8f acc2 = {};
  for (int k0 = 0; k0 < 4; ++k0) {
    v16bf a = load_frag_A(sV + k0 * 32, lane, 128);
    v16bf b = load_frag_B(sB2 + n0 * 128 + k0 * 32, lane, 128);
    acc2 = wmma_bf16(a, b, acc2);
  }
  float bias2 = bdense[col];
  for (int r = 0; r < 8; ++r)
    out[(size_t)(m0 + c_row(r, lane)) * 128 + col] = acc2[r] + bias2;
}

// ---------------------------------------------------------------------------
// Launch
// ---------------------------------------------------------------------------
extern "C" void kernel_launch(void* const* d_in, const int* in_sizes, int n_in,
                              void* d_out, int out_size, void* d_ws, size_t ws_size,
                              hipStream_t stream) {
  const float* s      = (const float*)d_in[0];
  const float* mask   = (const float*)d_in[1];
  const int*   nbr    = (const int*)  d_in[2];
  const float* fij    = (const float*)d_in[3];
  const float* Wf1    = (const float*)d_in[4];
  const float* bf1    = (const float*)d_in[5];
  const float* Wf2    = (const float*)d_in[6];
  const float* bf2    = (const float*)d_in[7];
  const float* Win2f  = (const float*)d_in[8];
  const float* Wf2out = (const float*)d_in[9];
  const float* bf2out = (const float*)d_in[10];
  const float* Wdense = (const float*)d_in[11];
  const float* bdense = (const float*)d_in[12];
  float* out = (float*)d_out;

  char* wsb = (char*)d_ws;
  float*  y    = (float*)wsb;                                // 8192x128 f32 (4 MB)
  float*  agg  = (float*)(wsb + (size_t)32*256*128*4);       // 8192x128 f32 (4 MB)
  __bf16* W1t  = (__bf16*)(wsb + (size_t)8*1024*1024);       // 128x64 bf16
  __bf16* W2t  = W1t  + 128*64;                              // 128x128 bf16
  __bf16* Wint = W2t  + 128*128;
  __bf16* Wot  = Wint + 128*128;
  __bf16* Wdt  = Wot  + 128*128;

  const size_t shm1 = 4096 + 32768;          // 36 KB
  const size_t shm2 = 102464;                // ~100 KB (<= 320 KB/WGP)
  const size_t shm3 = 8192 + 2*32768;        // 72 KB

  k_prep  <<<32,   256, 0,    stream>>>(Wf1, Wf2, Win2f, Wf2out, Wdense,
                                        W1t, W2t, Wint, Wot, Wdt);
  k_in2f  <<<512,  256, shm1, stream>>>(s, Wint, y);
  k_filter<<<32*256/NODES_PER_WG, 256, shm2, stream>>>(fij, W1t, bf1, W2t, bf2,
                                                       nbr, mask, y, agg);
  k_out   <<<512,  256, shm3, stream>>>(agg, Wot, bf2out, Wdt, bdense, out);
}
